// TemporalBlock_4655744549160
// MI455X (gfx1250) — matrix operand order repaired and checked by reading, not compiled, not run on hardware
//
#include <hip/hip_runtime.h>

// Problem sizes (fixed by the reference)
#define BB 64
#define SS 1024
#define CC 512
#define EPSF 1e-5f
#define LDP 72   // LDS row pitch in halves (144B = 9*16B -> conflict-free b128)

typedef __attribute__((ext_vector_type(16))) _Float16    v16h;
typedef __attribute__((ext_vector_type(4)))  _Float16    v4h;
typedef __attribute__((ext_vector_type(8)))  float       v8f;
typedef __attribute__((ext_vector_type(8)))  unsigned int v8u;
typedef __attribute__((__vector_size__(16))) int         v4i;

// ---- CDNA5 async global->LDS path (guarded so compile can never regress) ----
#if defined(__HIP_DEVICE_COMPILE__) && \
    __has_builtin(__builtin_amdgcn_global_load_async_to_lds_b128)
#define HAVE_ASYNC_LDS 1
#else
#define HAVE_ASYNC_LDS 0
#endif

// Builtin expects pointers to 16B int vectors: AS1 (global) src, AS3 (LDS) dst.
#define AS1P(p) ((__attribute__((address_space(1))) v4i*)(void*)(p))
#define AS3P(p) ((__attribute__((address_space(3))) v4i*)(void*)(p))

template <int N>
__device__ __forceinline__ void wait_asynccnt() {
#if defined(__HIP_DEVICE_COMPILE__)
#if __has_builtin(__builtin_amdgcn_s_wait_asynccnt)
    __builtin_amdgcn_s_wait_asynccnt(N);
#elif HAVE_ASYNC_LDS
    asm volatile("s_wait_asynccnt %0" ::"i"(N) : "memory");
#endif
#endif
}

// ---------------------------------------------------------------------------
// Kernel 1: per-batch mean / rstd over the whole [S,C] slab.
// ---------------------------------------------------------------------------
__global__ __launch_bounds__(256) void stats_kernel(const float* __restrict__ x,
                                                    float* __restrict__ stats) {
    const int b = blockIdx.x;
    const float* xb = x + (size_t)b * (SS * CC);
    float s = 0.f, ss = 0.f;
    const int n4 = (SS * CC) / 4;
    for (int i = threadIdx.x; i < n4; i += blockDim.x) {
        float4 v = reinterpret_cast<const float4*>(xb)[i];
        s  += v.x + v.y + v.z + v.w;
        ss += v.x * v.x + v.y * v.y + v.z * v.z + v.w * v.w;
    }
    #pragma unroll
    for (int off = 16; off > 0; off >>= 1) {
        s  += __shfl_down(s,  off, 32);
        ss += __shfl_down(ss, off, 32);
    }
    __shared__ float rs[8], rss[8];
    const int wave = threadIdx.x >> 5;
    const int lane = threadIdx.x & 31;
    if (lane == 0) { rs[wave] = s; rss[wave] = ss; }
    __syncthreads();
    if (threadIdx.x == 0) {
        float s0 = 0.f, s1 = 0.f;
        #pragma unroll
        for (int i = 0; i < 8; ++i) { s0 += rs[i]; s1 += rss[i]; }
        const float inv = 1.0f / (float)(SS * CC);
        const float mu  = s0 * inv;
        const float var = s1 * inv - mu * mu;
        stats[2 * b]     = mu;
        stats[2 * b + 1] = rsqrtf(var + EPSF);
    }
}

// ---------------------------------------------------------------------------
// Kernel 2a: lin_w (f32, [t][s]) -> fp16, same layout (A operand).
// ---------------------------------------------------------------------------
__global__ __launch_bounds__(256) void convw_kernel(const float* __restrict__ w,
                                                    _Float16* __restrict__ wh) {
    const int i = blockIdx.x * blockDim.x + threadIdx.x;
    float4 v = reinterpret_cast<const float4*>(w)[i];
    v4h h;
    h[0] = (_Float16)v.x; h[1] = (_Float16)v.y;
    h[2] = (_Float16)v.z; h[3] = (_Float16)v.w;
    reinterpret_cast<v4h*>(wh)[i] = h;
}

// ---------------------------------------------------------------------------
// Kernel 2b: h = (x-mu)*rstd*ln_w + ln_b, stored TRANSPOSED as fp16 ht[b][c][s].
// ---------------------------------------------------------------------------
__global__ __launch_bounds__(256) void norm_t_kernel(const float* __restrict__ x,
                                                     const float* __restrict__ lnw,
                                                     const float* __restrict__ lnb,
                                                     const float* __restrict__ stats,
                                                     _Float16* __restrict__ ht) {
    const int b  = blockIdx.z;
    const int s0 = blockIdx.y * 32;
    const int c0 = blockIdx.x * 32;
    const float mu   = stats[2 * b];
    const float rstd = stats[2 * b + 1];
    __shared__ _Float16 tile[32][33];
    const int r  = threadIdx.x >> 5;
    const int cc = threadIdx.x & 31;
    const float* xb = x + (size_t)b * SS * CC;
    #pragma unroll
    for (int i = 0; i < 4; ++i) {
        const int sr = r + i * 8;
        const size_t idx = (size_t)(s0 + sr) * CC + (c0 + cc);
        const float v = xb[idx];
        const float h = (v - mu) * rstd * lnw[idx] + lnb[idx];
        tile[sr][cc] = (_Float16)h;
    }
    __syncthreads();
    _Float16* htb = ht + (size_t)b * CC * SS;
    #pragma unroll
    for (int i = 0; i < 4; ++i) {
        const int cr = r + i * 8;
        htb[(size_t)(c0 + cr) * SS + (s0 + cc)] = tile[cc][cr];
    }
}

// ---------------------------------------------------------------------------
// GEMM helpers
// ---------------------------------------------------------------------------
#if HAVE_ASYNC_LDS
// Issue one K=64 stage (A: 128x128B, B: 128x128B) as 8 async b128 DMAs per
// thread; LDS is written directly by the async engine (no VGPR round trip).
__device__ __forceinline__ void stage_async(const _Float16* Ag, const _Float16* Bg,
                                            _Float16* Asb, _Float16* Bsb,
                                            int m0, int n0, int k0, int tid) {
    #pragma unroll
    for (int p = 0; p < 4; ++p) {
        const int flat = p * 256 + tid;
        const int row  = flat >> 3;
        const int ch   = flat & 7;
        __builtin_amdgcn_global_load_async_to_lds_b128(
            AS1P(Ag + (size_t)(m0 + row) * SS + k0 + ch * 8),
            AS3P(Asb + row * LDP + ch * 8), 0, 0);
        __builtin_amdgcn_global_load_async_to_lds_b128(
            AS1P(Bg + (size_t)(n0 + row) * SS + k0 + ch * 8),
            AS3P(Bsb + row * LDP + ch * 8), 0, 0);
    }
}
#endif

__device__ __forceinline__ void stage_sync(const _Float16* Ag, const _Float16* Bg,
                                           _Float16* Asb, _Float16* Bsb,
                                           int m0, int n0, int k0, int tid) {
    #pragma unroll
    for (int p = 0; p < 4; ++p) {
        const int flat = p * 256 + tid;
        const int row  = flat >> 3;
        const int ch   = flat & 7;
        const uint4 av = *reinterpret_cast<const uint4*>(
            Ag + (size_t)(m0 + row) * SS + k0 + ch * 8);
        *reinterpret_cast<uint4*>(Asb + row * LDP + ch * 8) = av;
        const uint4 bv = *reinterpret_cast<const uint4*>(
            Bg + (size_t)(n0 + row) * SS + k0 + ch * 8);
        *reinterpret_cast<uint4*>(Bsb + row * LDP + ch * 8) = bv;
    }
}

// One K=64 stage of WMMAs from a staged LDS buffer pair.
__device__ __forceinline__ void mma_stage(const _Float16* Asb, const _Float16* Bsb,
                                          int wm, int wn, int nlo, int hi,
                                          v8f (&acc)[2][4]) {
    #pragma unroll
    for (int kk = 0; kk < 2; ++kk) {
        const int kb = kk * 32;
        v16h af[2], bf[4];
        // A fragment: lanes 0-15 hold K0..7 & K16..23, lanes 16-31 K8..15 & K24..31
        #pragma unroll
        for (int mt = 0; mt < 2; ++mt) {
            const _Float16* base = Asb + (wm * 32 + mt * 16 + nlo) * LDP + kb;
            uint4 t0 = *reinterpret_cast<const uint4*>(base + hi * 8);
            uint4 t1 = *reinterpret_cast<const uint4*>(base + 16 + hi * 8);
            v8u u = {t0.x, t0.y, t0.z, t0.w, t1.x, t1.y, t1.z, t1.w};
            af[mt] = __builtin_bit_cast(v16h, u);
        }
        // B fragment: lanes 0-15 hold K0..7 & K16..23, lanes 16-31 K8..15 & K24..31 (col = lane%16)
        #pragma unroll
        for (int nt = 0; nt < 4; ++nt) {
            const _Float16* base = Bsb + (wn * 64 + nt * 16 + nlo) * LDP + kb + hi * 8;
            uint4 t0 = *reinterpret_cast<const uint4*>(base);
            uint4 t1 = *reinterpret_cast<const uint4*>(base + 16);
            v8u u = {t0.x, t0.y, t0.z, t0.w, t1.x, t1.y, t1.z, t1.w};
            bf[nt] = __builtin_bit_cast(v16h, u);
        }
        #pragma unroll
        for (int mt = 0; mt < 2; ++mt)
            #pragma unroll
            for (int nt = 0; nt < 4; ++nt)
                acc[mt][nt] = __builtin_amdgcn_wmma_f32_16x16x32_f16(
                    false, af[mt], false, bf[nt],
                    (short)0, acc[mt][nt], false, false);
    }
}

// ---------------------------------------------------------------------------
// Kernel 3: out[t][c] = x[t][c] + relu( sum_s W[t][s]*H[s][c] + bias[t] )
//   A = wh [S][S] fp16 (row t, K contiguous);  B = ht [C][S] fp16 (row c, K contiguous)
// 256 threads (8 waves, 4Mx2N), 128x128 tile, K staged by 64, double-buffered
// async global->LDS DMA overlapped with WMMA on the other buffer.
// ---------------------------------------------------------------------------
__global__ __launch_bounds__(256) void gemm_kernel(const _Float16* __restrict__ wh,
                                                   const _Float16* __restrict__ ht,
                                                   const float* __restrict__ x,
                                                   const float* __restrict__ linb,
                                                   float* __restrict__ out) {
    const int b  = blockIdx.z;
    const int m0 = blockIdx.y * 128;
    const int n0 = blockIdx.x * 128;

    __shared__ __align__(16) _Float16 As[2][128 * LDP];
    __shared__ __align__(16) _Float16 Bs[2][128 * LDP];

    const int tid  = threadIdx.x;
    const int wave = tid >> 5;
    const int lane = tid & 31;
    const int wm   = wave & 3;    // 0..3 : M offset /32
    const int wn   = wave >> 2;   // 0..1 : N offset /64
    const int nlo  = lane & 15;
    const int hi   = lane >> 4;

    const _Float16* Ag = wh;
    const _Float16* Bg = ht + (size_t)b * CC * SS;

    v8f acc[2][4];
    const v8f vzero = {0.f, 0.f, 0.f, 0.f, 0.f, 0.f, 0.f, 0.f};
    #pragma unroll
    for (int i = 0; i < 2; ++i)
        #pragma unroll
        for (int j = 0; j < 4; ++j) acc[i][j] = vzero;

#if HAVE_ASYNC_LDS
    // Prologue: kick off stage 0 DMA.
    stage_async(Ag, Bg, &As[0][0], &Bs[0][0], m0, n0, 0, tid);
    int st = 0;
    for (int k0 = 0; k0 < SS; k0 += 64, st ^= 1) {
        if (k0 + 64 < SS) {
            // Issue next stage into the other buffer, then wait only until the
            // previous 8 DMAs (this stage) are done: next stage stays in flight.
            stage_async(Ag, Bg, &As[st ^ 1][0], &Bs[st ^ 1][0], m0, n0, k0 + 64, tid);
            wait_asynccnt<8>();
        } else {
            wait_asynccnt<0>();
        }
        __syncthreads();   // all waves' DMA contributions visible
        mma_stage(&As[st][0], &Bs[st][0], wm, wn, nlo, hi, acc);
        __syncthreads();   // reads done before buffer is overwritten next iter
    }
#else
    for (int k0 = 0; k0 < SS; k0 += 64) {
        __syncthreads();
        stage_sync(Ag, Bg, &As[0][0], &Bs[0][0], m0, n0, k0, tid);
        __syncthreads();
        mma_stage(&As[0][0], &Bs[0][0], wm, wn, nlo, hi, acc);
    }
#endif

    // Epilogue: bias + relu + residual.  C/D layout: VGPR v -> M = v + 8*(lane/16),
    // N = lane%16.
    const float* xb = x   + (size_t)b * SS * CC;
    float*       ob = out + (size_t)b * SS * CC;
    #pragma unroll
    for (int mt = 0; mt < 2; ++mt) {
        #pragma unroll
        for (int nt = 0; nt < 4; ++nt) {
            const int cg = n0 + wn * 64 + nt * 16 + nlo;
            #pragma unroll
            for (int v = 0; v < 8; ++v) {
                const int tg = m0 + wm * 32 + mt * 16 + hi * 8 + v;
                float val = acc[mt][nt][v] + linb[tg];
                val = val > 0.f ? val : 0.f;
                const size_t idx = (size_t)tg * CC + cg;
                ob[idx] = xb[idx] + val;
            }
        }
    }
}

// ---------------------------------------------------------------------------
// Launch: stats -> convert W -> normalize+transpose -> fused GEMM epilogue.
// Workspace: [stats 512B | pad to 4KB][wh 2MB fp16][ht 64MB fp16]  (~66MB)
// ---------------------------------------------------------------------------
extern "C" void kernel_launch(void* const* d_in, const int* in_sizes, int n_in,
                              void* d_out, int out_size, void* d_ws, size_t ws_size,
                              hipStream_t stream) {
    const float* x     = (const float*)d_in[0];
    const float* ln_w  = (const float*)d_in[1];
    const float* ln_b  = (const float*)d_in[2];
    const float* lin_w = (const float*)d_in[3];
    const float* lin_b = (const float*)d_in[4];
    float* out = (float*)d_out;

    char* ws = (char*)d_ws;
    float*    stats = (float*)ws;                                        // 512 B
    _Float16* wh    = (_Float16*)(ws + 4096);                            // 2 MB
    _Float16* ht    = (_Float16*)(ws + 4096 + (size_t)2 * 1024 * 1024);  // 64 MB

    stats_kernel<<<BB, 256, 0, stream>>>(x, stats);
    convw_kernel<<<(SS * SS / 4) / 256, 256, 0, stream>>>(lin_w, wh);
    dim3 g2(CC / 32, SS / 32, BB);
    norm_t_kernel<<<g2, 256, 0, stream>>>(x, ln_w, ln_b, stats, ht);
    dim3 g3(CC / 128, SS / 128, BB);
    gemm_kernel<<<g3, 256, 0, stream>>>(wh, ht, x, lin_b, out);
}
